// MultiHeadAttention_35957466202155
// MI455X (gfx1250) — matrix operand order repaired and checked by reading, not compiled, not run on hardware
//
#include <hip/hip_runtime.h>

#define D_MODEL 1024
#define NHEADS  16
#define DK      64
#define BATCH   4
#define SEQ     2048
#define MTOK    (BATCH * SEQ)   // 8192 token rows

typedef __attribute__((ext_vector_type(16))) __bf16 v16bf;
typedef __attribute__((ext_vector_type(8)))  __bf16 v8bf;
typedef __attribute__((ext_vector_type(8)))  float  v8f;

// ---------- helpers ----------
__device__ __forceinline__ unsigned short f2bf(float f) {
  unsigned u = __float_as_uint(f);
  u += 0x7FFFu + ((u >> 16) & 1u);       // round-to-nearest-even
  return (unsigned short)(u >> 16);
}
__device__ __forceinline__ unsigned pk2(float a, float b) {
  return (unsigned)f2bf(a) | ((unsigned)f2bf(b) << 16);
}
__device__ __forceinline__ v8bf ld8(const unsigned short* p) {
  return *reinterpret_cast<const v8bf*>(p);
}
__device__ __forceinline__ v16bf cat16(v8bf lo, v8bf hi) {
  return __builtin_shufflevector(lo, hi, 0,1,2,3,4,5,6,7,8,9,10,11,12,13,14,15);
}
__device__ __forceinline__ v8f wmma_bf16(v16bf a, v16bf b, v8f c) {
  // (neg_a, A, neg_b, B, c_mod, C, reuse_a, reuse_b)
  return __builtin_amdgcn_wmma_f32_16x16x32_bf16(false, a, false, b, (short)0, c, false, false);
}
__device__ __forceinline__ v8f zero8() {
  v8f z;
  #pragma unroll
  for (int i = 0; i < 8; ++i) z[i] = 0.0f;
  return z;
}

// ---------- 16-lane row reductions (VALU DPP, no LDS) ----------
// Rows of the WMMA C layout live inside one 16-lane half, and DPP16 row ops
// never cross 16-lane rows. xor-1, xor-2, xor-7, xor-15 are 4 independent
// generators of the 16-lane xor group -> full butterfly in 4 VALU ops.
#if __has_builtin(__builtin_amdgcn_update_dpp)
template<int CTRL>
__device__ __forceinline__ float dppf(float v) {
  return __int_as_float(
      __builtin_amdgcn_update_dpp(0, __float_as_int(v), CTRL, 0xF, 0xF, true));
}
__device__ __forceinline__ float redmax16(float v) {
  v = fmaxf(v, dppf<0xB1>(v));    // quad_perm [1,0,3,2]  (xor 1)
  v = fmaxf(v, dppf<0x4E>(v));    // quad_perm [2,3,0,1]  (xor 2)
  v = fmaxf(v, dppf<0x141>(v));   // row_half_mirror      (xor 7)
  v = fmaxf(v, dppf<0x140>(v));   // row_mirror           (xor 15)
  return v;
}
__device__ __forceinline__ float redsum16(float v) {
  v += dppf<0xB1>(v);
  v += dppf<0x4E>(v);
  v += dppf<0x141>(v);
  v += dppf<0x140>(v);
  return v;
}
#else
__device__ __forceinline__ float redmax16(float v) {
  #pragma unroll
  for (int off = 1; off < 16; off <<= 1) v = fmaxf(v, __shfl_xor(v, off, 16));
  return v;
}
__device__ __forceinline__ float redsum16(float v) {
  #pragma unroll
  for (int off = 1; off < 16; off <<= 1) v += __shfl_xor(v, off, 16);
  return v;
}
#endif

// ---------------------------------------------------------------------------
// Generic GEMM: C[M=8192, N=1024] = X[M,K=1024] @ W[N,K]^T + bias[N]
// MODE 0: store bf16 split-head  [B,H,S,Dk]           (Q, K projections)
// MODE 1: store bf16 transposed  [B,H,Dk,S]           (V projection)
// MODE 2: store fp32 flat        [M,N]                (output projection)
// XF32:   X is fp32 (true) or bf16/ushort (false)
// Block: 256 thr (8 waves), tile 128x128, K-step 32, double-buffered LDS.
// ---------------------------------------------------------------------------
template<int MODE, bool XF32>
__global__ __launch_bounds__(256)
void gemm_bias_kernel(const void* __restrict__ Xv,
                      const float* __restrict__ W,
                      const float* __restrict__ bias,
                      void* __restrict__ Out)
{
  __shared__ __align__(16) unsigned short As[2][128][40];  // pad 40: bank spread
  __shared__ __align__(16) unsigned short Bs[2][128][40];

  const int tid  = threadIdx.x;
  const int lane = tid & 31;
  const int w    = tid >> 5;
  const int half = lane >> 4;     // 0: rows M+0..7 / K-lo   1: rows M+8..15 / K-hi
  const int l16  = lane & 15;
  const int wm   = w & 1;         // 2 waves along M
  const int wn   = w >> 1;        // 4 waves along N
  const int m0   = blockIdx.y * 128;
  const int n0   = blockIdx.x * 128;

  const int srow = tid >> 1;            // staging row 0..127
  const int scol = (tid & 1) * 16;      // staging col 0/16

  v8f acc[4][2];
  #pragma unroll
  for (int mi = 0; mi < 4; ++mi)
    #pragma unroll
    for (int ni = 0; ni < 2; ++ni)
      acc[mi][ni] = zero8();

  auto stage = [&](int buf, int kk) {
    if constexpr (XF32) {
      const float* X = (const float*)Xv;
      const float4* px = (const float4*)(X + (size_t)(m0 + srow) * D_MODEL + kk + scol);
      float4 x0 = px[0], x1 = px[1], x2 = px[2], x3 = px[3];
      uint4 u0, u1;
      u0.x = pk2(x0.x, x0.y); u0.y = pk2(x0.z, x0.w);
      u0.z = pk2(x1.x, x1.y); u0.w = pk2(x1.z, x1.w);
      u1.x = pk2(x2.x, x2.y); u1.y = pk2(x2.z, x2.w);
      u1.z = pk2(x3.x, x3.y); u1.w = pk2(x3.z, x3.w);
      *(uint4*)&As[buf][srow][scol]     = u0;
      *(uint4*)&As[buf][srow][scol + 8] = u1;
      if (kk + 64 < D_MODEL)
        __builtin_prefetch(X + (size_t)(m0 + srow) * D_MODEL + kk + 64 + scol, 0, 1);
    } else {
      const unsigned short* X = (const unsigned short*)Xv;
      const uint4* px = (const uint4*)(X + (size_t)(m0 + srow) * D_MODEL + kk + scol);
      *(uint4*)&As[buf][srow][scol]     = px[0];
      *(uint4*)&As[buf][srow][scol + 8] = px[1];
      if (kk + 64 < D_MODEL)
        __builtin_prefetch(X + (size_t)(m0 + srow) * D_MODEL + kk + 64 + scol, 0, 1);
    }
    {
      const float4* pw = (const float4*)(W + (size_t)(n0 + srow) * D_MODEL + kk + scol);
      float4 x0 = pw[0], x1 = pw[1], x2 = pw[2], x3 = pw[3];
      uint4 u0, u1;
      u0.x = pk2(x0.x, x0.y); u0.y = pk2(x0.z, x0.w);
      u0.z = pk2(x1.x, x1.y); u0.w = pk2(x1.z, x1.w);
      u1.x = pk2(x2.x, x2.y); u1.y = pk2(x2.z, x2.w);
      u1.z = pk2(x3.x, x3.y); u1.w = pk2(x3.z, x3.w);
      *(uint4*)&Bs[buf][srow][scol]     = u0;
      *(uint4*)&Bs[buf][srow][scol + 8] = u1;
      if (kk + 64 < D_MODEL)
        __builtin_prefetch(W + (size_t)(n0 + srow) * D_MODEL + kk + 64 + scol, 0, 1);
    }
  };

  stage(0, 0);
  int buf = 0;
  for (int kk = 0; kk < D_MODEL; kk += 32) {
    __syncthreads();                       // staged buf visible / prev reads done
    if (kk + 32 < D_MODEL) stage(buf ^ 1, kk + 32);

    v16bf afr[4];
    #pragma unroll
    for (int mi = 0; mi < 4; ++mi) {
      const unsigned short* p = &As[buf][wm*64 + mi*16 + l16][half * 16];
      afr[mi] = cat16(ld8(p), ld8(p + 8));    // K 0-15 (lo) / 16-31 (hi)
    }
    v16bf bfr[2];
    #pragma unroll
    for (int ni = 0; ni < 2; ++ni) {
      const unsigned short* p = &Bs[buf][wn*32 + ni*16 + l16][half * 16];
      bfr[ni] = cat16(ld8(p), ld8(p + 8));    // 16 contiguous K of column l16
    }
    #pragma unroll
    for (int mi = 0; mi < 4; ++mi)
      #pragma unroll
      for (int ni = 0; ni < 2; ++ni)
        acc[mi][ni] = wmma_bf16(afr[mi], bfr[ni], acc[mi][ni]);
    buf ^= 1;
  }

  // ---- epilogue: bias + layout-specific store ----
  #pragma unroll
  for (int mi = 0; mi < 4; ++mi) {
    #pragma unroll
    for (int ni = 0; ni < 2; ++ni) {
      const int n_g = n0 + wn*32 + ni*16 + l16;
      const float bv = bias[n_g];
      #pragma unroll
      for (int r = 0; r < 8; ++r) {
        const int m_g = m0 + wm*64 + mi*16 + r + half*8;  // C layout: row = r + 8*half
        const float val = acc[mi][ni][r] + bv;
        if constexpr (MODE == 2) {
          ((float*)Out)[(size_t)m_g * D_MODEL + n_g] = val;
        } else {
          const int b = m_g >> 11, s = m_g & (SEQ - 1);
          const int h = n_g >> 6,  d = n_g & (DK - 1);
          size_t idx;
          if constexpr (MODE == 0) idx = (((size_t)(b*NHEADS + h)) * SEQ + s) * DK + d;
          else                     idx = (((size_t)(b*NHEADS + h)) * DK + d) * SEQ + s;
          ((unsigned short*)Out)[idx] = f2bf(val);
        }
      }
    }
  }
}

// ---------------------------------------------------------------------------
// Flash attention per (b,h). Q,K bf16 [B,H,S,Dk]; V bf16 transposed [B,H,Dk,S].
// Block = 128 thr (4 waves); each wave owns 16 query rows; keys in tiles of 64
// (16 WMMAs per softmax update). Reference quirk: softmax on UNSCALED scores,
// divide by sqrt(Dk)=8 AFTER softmax -> folded into final 1/(8*l) scale.
// ---------------------------------------------------------------------------
__global__ __launch_bounds__(128)
void attn_kernel(const unsigned short* __restrict__ Qb,
                 const unsigned short* __restrict__ Kb,
                 const unsigned short* __restrict__ Vt,
                 unsigned short* __restrict__ AO)
{
  __shared__ __align__(16) unsigned short Plds[4][16][64];  // per-wave P tile (8KB)

  const int tid  = threadIdx.x;
  const int lane = tid & 31;
  const int w    = tid >> 5;
  const int half = lane >> 4;
  const int l16  = lane & 15;

  const int bh = blockIdx.y;                 // 0..63
  const int b  = bh >> 4, h = bh & 15;
  const int q0 = blockIdx.x * 64 + w * 16;

  const unsigned short* Qg = Qb + (size_t)bh * SEQ * DK;
  const unsigned short* Kg = Kb + (size_t)bh * SEQ * DK;
  const unsigned short* Vg = Vt + (size_t)bh * DK * SEQ;

  // Q fragments: 16 rows x Dk=64 -> two 16x32 A-frags, contiguous loads
  v16bf qf[2];
  #pragma unroll
  for (int kd = 0; kd < 2; ++kd) {
    const unsigned short* p = Qg + (size_t)(q0 + l16) * DK + kd*32 + half*16;
    qf[kd] = cat16(ld8(p), ld8(p + 8));
  }

  float m_run[8], l_run[8];
  v8f o[4];
  #pragma unroll
  for (int r = 0; r < 8; ++r) { m_run[r] = -INFINITY; l_run[r] = 0.0f; }
  #pragma unroll
  for (int dn = 0; dn < 4; ++dn) o[dn] = zero8();

  for (int kt = 0; kt < SEQ; kt += 64) {
    // ---- scores for 64 keys: four 16x16 tiles (8 WMMAs) ----
    v8f sc[4];
    #pragma unroll
    for (int ni = 0; ni < 4; ++ni) {
      v8f z = zero8();
      #pragma unroll
      for (int kd = 0; kd < 2; ++kd) {
        // B[d][n] = K[key_n][d]: 16 contiguous d of key row -> contiguous load
        const unsigned short* p = Kg + (size_t)(kt + ni*16 + l16) * DK + kd*32 + half*16;
        v16bf kf = cat16(ld8(p), ld8(p + 8));
        z = wmma_bf16(qf[kd], kf, z);
      }
      sc[ni] = z;
    }

    // ---- online softmax (lanes 0-15 hold rows r, lanes 16-31 rows r+8) ----
    float mt[8], alpha[8];
    #pragma unroll
    for (int r = 0; r < 8; ++r) {
      float v = fmaxf(fmaxf(sc[0][r], sc[1][r]), fmaxf(sc[2][r], sc[3][r]));
      v = redmax16(v);
      mt[r]    = fmaxf(m_run[r], v);
      alpha[r] = __expf(m_run[r] - mt[r]);
      m_run[r] = mt[r];
    }

    __syncthreads();
    #pragma unroll
    for (int r = 0; r < 8; ++r) {
      float p0 = __expf(sc[0][r] - mt[r]);
      float p1 = __expf(sc[1][r] - mt[r]);
      float p2 = __expf(sc[2][r] - mt[r]);
      float p3 = __expf(sc[3][r] - mt[r]);
      Plds[w][r + half*8][l16]      = f2bf(p0);   // re-layout C -> A via LDS
      Plds[w][r + half*8][16 + l16] = f2bf(p1);
      Plds[w][r + half*8][32 + l16] = f2bf(p2);
      Plds[w][r + half*8][48 + l16] = f2bf(p3);
      float s = redsum16((p0 + p1) + (p2 + p3));
      l_run[r] = l_run[r] * alpha[r] + s;
    }
    __syncthreads();

    // ---- rescale O, then O += P @ V (8 WMMAs) ----
    #pragma unroll
    for (int dn = 0; dn < 4; ++dn)
      #pragma unroll
      for (int r = 0; r < 8; ++r)
        o[dn][r] *= alpha[r];

    v16bf pf[2];
    #pragma unroll
    for (int ks = 0; ks < 2; ++ks) {
      const unsigned short* pp = &Plds[w][l16][ks*32 + half*16];
      pf[ks] = cat16(ld8(pp), ld8(pp + 8));       // P as 16x32 A-frags
    }
    #pragma unroll
    for (int dn = 0; dn < 4; ++dn) {
      #pragma unroll
      for (int ks = 0; ks < 2; ++ks) {
        // V transposed [Dk,S]: B[key][d] frag = 16 contiguous keys of row d
        const unsigned short* pv = Vg + (size_t)(dn*16 + l16) * SEQ + kt + ks*32 + half*16;
        v16bf vf = cat16(ld8(pv), ld8(pv + 8));
        o[dn] = wmma_bf16(pf[ks], vf, o[dn]);
      }
    }
  }

  // ---- normalize (1/l) and post-softmax 1/sqrt(Dk)=1/8; merged-head store ----
  #pragma unroll
  for (int r = 0; r < 8; ++r) {
    const float inv = 1.0f / (l_run[r] * 8.0f);
    const int s = q0 + r + half*8;
    #pragma unroll
    for (int dn = 0; dn < 4; ++dn) {
      const int d = dn*16 + l16;
      AO[((size_t)(b * SEQ + s)) * D_MODEL + h*DK + d] = f2bf(o[dn][r] * inv);
    }
  }
}

// ---------------------------------------------------------------------------
extern "C" void kernel_launch(void* const* d_in, const int* in_sizes, int n_in,
                              void* d_out, int out_size, void* d_ws, size_t ws_size,
                              hipStream_t stream) {
  (void)in_sizes; (void)n_in; (void)out_size; (void)ws_size;
  const float* q  = (const float*)d_in[0];
  const float* k  = (const float*)d_in[1];
  const float* v  = (const float*)d_in[2];
  const float* WQ = (const float*)d_in[3];
  const float* bQ = (const float*)d_in[4];
  const float* WK = (const float*)d_in[5];
  const float* bK = (const float*)d_in[6];
  const float* WV = (const float*)d_in[7];
  const float* bV = (const float*)d_in[8];
  const float* W0 = (const float*)d_in[9];
  const float* b0 = (const float*)d_in[10];

  // bf16 workspace: Q,K [B,H,S,Dk] ; V transposed [B,H,Dk,S] ; attn-out [M,D]
  unsigned short* Qb = (unsigned short*)d_ws;
  unsigned short* Kb = Qb + (size_t)MTOK * D_MODEL;
  unsigned short* Vt = Kb + (size_t)MTOK * D_MODEL;
  unsigned short* AO = Vt + (size_t)MTOK * D_MODEL;

  const dim3 gg(D_MODEL / 128, MTOK / 128);   // (8, 64) blocks, 256 thr each
  gemm_bias_kernel<0, true ><<<gg, 256, 0, stream>>>(q,  WQ, bQ, Qb);
  gemm_bias_kernel<0, true ><<<gg, 256, 0, stream>>>(k,  WK, bK, Kb);
  gemm_bias_kernel<1, true ><<<gg, 256, 0, stream>>>(v,  WV, bV, Vt);
  attn_kernel<<<dim3(SEQ / 64, BATCH * NHEADS), 128, 0, stream>>>(Qb, Kb, Vt, AO);
  gemm_bias_kernel<2, false><<<gg, 256, 0, stream>>>(AO, W0, b0, (float*)d_out);
}
